// GCNClassifier_90529320665178
// MI455X (gfx1250) — matrix operand order repaired and checked by reading, not compiled, not run on hardware
//
#include <hip/hip_runtime.h>

// ---------------- types for WMMA ----------------
typedef _Float16 h16;
typedef __attribute__((ext_vector_type(16))) _Float16 v16h;
typedef __attribute__((ext_vector_type(8)))  _Float16 v8h;
typedef __attribute__((ext_vector_type(8)))  float    v8f;

#define B_    16
#define T_    1024
#define D_    500
#define DP    512            // D padded to WMMA tiles
#define NROW  (B_ * T_)      // 16384
#define SCALE 22.360679775f  // sqrt(H2=500)
#define NEGBIG 1e12f

// A fragment: 16x32 f16, row-major source. lane lo holds row M=lo,
// elements i -> K = k0+16*hi+i.
__device__ __forceinline__ v16h frag_a(const h16* row0, int lda, int k0, int lane) {
  const int lo = lane & 15, hi = lane >> 4;
  return *(const v16h*)(row0 + (size_t)lo * lda + k0 + 16 * hi);
}
// B fragment: 32x16 f16 from a "Bt" matrix stored N-major (row n holds K
// contiguously). lane lo holds col N=lo; elements i -> K = k0+16*hi+i.
__device__ __forceinline__ v16h frag_bt(const h16* bt_row0, int ldb, int k0, int lane) {
  const int lo = lane & 15, hi = lane >> 4;
  return *(const v16h*)(bt_row0 + (size_t)lo * ldb + k0 + 16 * hi);
}
#define WMMA(a, b, c) __builtin_amdgcn_wmma_f32_16x16x32_f16(false, (a), false, (b), (short)0, (c), false, false)

// ---------------- embedding concat -> g16 (NROW x DP, f16, zero-padded) ----
__global__ void embed_kernel(const int* __restrict__ words, const int* __restrict__ pos,
                             const int* __restrict__ ner, const int* __restrict__ subj_pos,
                             const int* __restrict__ obj_pos, const int* __restrict__ chunks,
                             const int* __restrict__ on_path, const float* __restrict__ dep_feat,
                             const float* __restrict__ emb_w, const float* __restrict__ pos_w,
                             const float* __restrict__ ner_w, const float* __restrict__ chunk_w,
                             const float* __restrict__ position_w, h16* __restrict__ g16) {
  int idx = blockIdx.x * blockDim.x + threadIdx.x;
  if (idx >= NROW * DP) return;
  int row = idx >> 9, d = idx & 511;
  float v = 0.f;
  if      (d < 300) v = emb_w[(size_t)words[row] * 300 + d];
  else if (d < 335) v = pos_w[pos[row] * 35 + (d - 300)];
  else if (d < 365) v = ner_w[ner[row] * 30 + (d - 335)];
  else if (d < 395) v = chunk_w[chunks[row] * 30 + (d - 365)];
  else if (d < 425) v = position_w[subj_pos[row] * 30 + (d - 395)];
  else if (d < 455) v = position_w[obj_pos[row] * 30 + (d - 425)];
  else if (d == 455) v = (float)on_path[row];
  else if (d < 500) v = dep_feat[(size_t)row * 44 + (d - 456)];
  g16[idx] = (h16)v;
}

// ---------------- weight transpose+pad: W(500x500) -> Wt(512x512) f16 ------
__global__ void wtrans_kernel(const float* __restrict__ W, h16* __restrict__ Wt) {
  int idx = blockIdx.x * blockDim.x + threadIdx.x;
  if (idx >= DP * DP) return;
  int n = idx >> 9, k = idx & 511;
  Wt[idx] = (h16)((n < D_ && k < D_) ? W[(size_t)k * D_ + n] : 0.f);
}

// ---------------- projection GEMM: out = g16 @ W + bias (f16 out) ----------
// grid (8, 128), block 256 (8 waves); wave -> 16x64 tile.
__global__ __launch_bounds__(256) void proj_kernel(const h16* __restrict__ A,
                                                   const h16* __restrict__ Wt,
                                                   const float* __restrict__ bias,
                                                   h16* __restrict__ out, int store_t) {
  const int lane = threadIdx.x & 31, wave = threadIdx.x >> 5;
  const int lo = lane & 15, hi = lane >> 4;
  const int m0 = blockIdx.y * 128 + wave * 16;
  const int n0 = blockIdx.x * 64;
  v8f acc[4] = {};
  for (int k0 = 0; k0 < DP; k0 += 32) {
    v16h a = frag_a(A + (size_t)m0 * DP, DP, k0, lane);
#pragma unroll
    for (int nt = 0; nt < 4; ++nt) {
      v16h b = frag_bt(Wt + (size_t)(n0 + nt * 16) * DP, DP, k0, lane);
      acc[nt] = WMMA(a, b, acc[nt]);
    }
  }
#pragma unroll
  for (int nt = 0; nt < 4; ++nt) {
    int n = n0 + nt * 16 + lo;
    float bn = (n < D_) ? bias[n] : 0.f;
#pragma unroll
    for (int j = 0; j < 8; ++j) {
      int m = m0 + j + 8 * hi;
      float v = acc[nt][j] + bn;
      if (!store_t) out[(size_t)m * DP + n] = (h16)v;                    // row-major
      else { int bb = m >> 10, t = m & 1023;                             // [b][d][t]
             out[((size_t)bb * DP + n) * T_ + t] = (h16)v; }
    }
  }
}

// ---------------- subj/obj max-pool ----------------------------------------
__global__ void pool_kernel(const int* __restrict__ subj_pos, const h16* __restrict__ g16,
                            float* __restrict__ subj) {
  int idx = blockIdx.x * blockDim.x + threadIdx.x;
  if (idx >= B_ * D_) return;
  int b = idx / D_, d = idx % D_;
  float mx = -3.4e38f;
  for (int t = 0; t < T_; ++t) {
    int row = b * T_ + t;
    float v = (subj_pos[row] != 0) ? -NEGBIG : (float)g16[(size_t)row * DP + d];
    mx = fmaxf(mx, v);
  }
  subj[idx] = mx;
}

// ---------------- q = relu([subj,subj] @ Wq + b) ---------------------------
__global__ void q_kernel(const float* __restrict__ subj, const float* __restrict__ Wq,
                         const float* __restrict__ Wqb, float* __restrict__ q) {
  int idx = blockIdx.x * blockDim.x + threadIdx.x;
  if (idx >= B_ * D_) return;
  int b = idx / D_, n = idx % D_;
  float s = Wqb[n];
  for (int d = 0; d < D_; ++d)
    s += subj[b * D_ + d] * (Wq[(size_t)d * D_ + n] + Wq[(size_t)(d + D_) * D_ + n]);
  q[idx] = fmaxf(s, 0.f);
}

// ---------------- tv = relu([q,0] @ Wc + b) --------------------------------
__global__ void tv_kernel(const float* __restrict__ q, const float* __restrict__ Wc,
                          const float* __restrict__ Wcb, float* __restrict__ tv) {
  int idx = blockIdx.x * blockDim.x + threadIdx.x;
  if (idx >= B_ * D_) return;
  int b = idx / D_, n = idx % D_;
  float s = Wcb[n];
  for (int d = 0; d < D_; ++d) s += q[b * D_ + d] * Wc[(size_t)d * D_ + n];
  tv[idx] = fmaxf(s, 0.f);
}

// ---------------- k_logits[b,t] = (tv*g[b,t]) . Wk + b ---------------------
__global__ void klog_kernel(const float* __restrict__ tv, const h16* __restrict__ g16,
                            const float* __restrict__ Wk, const float* __restrict__ Wkb,
                            float* __restrict__ klog) {
  int row = blockIdx.x * blockDim.x + threadIdx.x;
  if (row >= NROW) return;
  int b = row >> 10;
  float s = Wkb[0];
  for (int d = 0; d < D_; ++d)
    s += tv[b * D_ + d] * Wk[d] * (float)g16[(size_t)row * DP + d];
  klog[row] = s;
}

// ---------------- softmax over T (axis=1) ----------------------------------
__global__ __launch_bounds__(256) void ksm_kernel(const float* __restrict__ klog,
                                                  float* __restrict__ ksm) {
  __shared__ float red[256];
  int b = blockIdx.x, tid = threadIdx.x;
  float mx = -3.4e38f;
  for (int t = tid; t < T_; t += 256) mx = fmaxf(mx, klog[b * T_ + t]);
  red[tid] = mx; __syncthreads();
  for (int s = 128; s > 0; s >>= 1) { if (tid < s) red[tid] = fmaxf(red[tid], red[tid + s]); __syncthreads(); }
  mx = red[0]; __syncthreads();
  float sum = 0.f;
  for (int t = tid; t < T_; t += 256) sum += expf(klog[b * T_ + t] - mx);
  red[tid] = sum; __syncthreads();
  for (int s = 128; s > 0; s >>= 1) { if (tid < s) red[tid] += red[tid + s]; __syncthreads(); }
  float inv = 1.f / red[0];
  for (int t = tid; t < T_; t += 256) ksm[b * T_ + t] = expf(klog[b * T_ + t] - mx) * inv;
}

// ---------------- c[b,d] = sum_t ksm*g -------------------------------------
__global__ void c_kernel(const float* __restrict__ ksm, const h16* __restrict__ g16,
                         float* __restrict__ c) {
  int idx = blockIdx.x * blockDim.x + threadIdx.x;
  if (idx >= B_ * D_) return;
  int b = idx / D_, d = idx % D_;
  float s = 0.f;
  for (int t = 0; t < T_; ++t)
    s += ksm[b * T_ + t] * (float)g16[(size_t)(b * T_ + t) * DP + d];
  c[idx] = s;
}

// ---------------- m = relu([c,subj,subj] @ Wm + b) -------------------------
__global__ void m_kernel(const float* __restrict__ c, const float* __restrict__ subj,
                         const float* __restrict__ Wm, const float* __restrict__ Wmb,
                         float* __restrict__ m) {
  int idx = blockIdx.x * blockDim.x + threadIdx.x;
  if (idx >= B_ * D_) return;
  int b = idx / D_, n = idx % D_;
  float s = Wmb[n];
  for (int d = 0; d < D_; ++d) {
    s += c[b * D_ + d] * Wm[(size_t)d * D_ + n];
    s += subj[b * D_ + d] * (Wm[(size_t)(500 + d) * D_ + n] + Wm[(size_t)(1000 + d) * D_ + n]);
  }
  m[idx] = fmaxf(s, 0.f);
}

// ---------------- fused attention: S slice in LDS, softmax, att, P@V -------
// grid (T/16, B), block 128 (4 waves). LDS-resident 16x1024 S slice.
__global__ __launch_bounds__(128) void attn_kernel(const h16* __restrict__ key16,
                                                   const h16* __restrict__ query16,
                                                   const h16* __restrict__ valueT,
                                                   const float* __restrict__ mvec,
                                                   float* __restrict__ out,
                                                   float* __restrict__ att) {
  __shared__ float Sl[16 * 1024];   // 64 KB
  __shared__ h16   P16[16 * 1024];  // 32 KB
  __shared__ float red[16 * 8];
  __shared__ float rmax[16], rsum1[16], rsum2[16];

  const int b = blockIdx.y;
  const int t0 = blockIdx.x * 16;
  const int tid = threadIdx.x;
  const int lane = tid & 31, wave = tid >> 5;
  const int lo = lane & 15, hi = lane >> 4;

  const h16* keyb = key16 + (size_t)b * T_ * DP;
  const h16* qryb = query16 + (size_t)b * T_ * DP;

  // phase 1: S[t0..t0+15][:] = key rows @ query^T (query rows act as Bt rows)
  for (int nt = 0; nt < 16; ++nt) {
    int n0 = wave * 256 + nt * 16;
    v8f acc = {};
    for (int k0 = 0; k0 < DP; k0 += 32) {
      v16h a = frag_a(keyb + (size_t)t0 * DP, DP, k0, lane);
      v16h bb = frag_bt(qryb + (size_t)n0 * DP, DP, k0, lane);
      acc = WMMA(a, bb, acc);
    }
#pragma unroll
    for (int j = 0; j < 8; ++j) Sl[(j + 8 * hi) * 1024 + n0 + lo] = acc[j];
  }
  __syncthreads();

  // phase 2: row stats (max, sum exp, sum exp(/scale)), diagonal att
  {
    int r = tid & 15, part = tid >> 4;
    float mx = -3.4e38f;
    for (int s = part * 128; s < part * 128 + 128; ++s) mx = fmaxf(mx, Sl[r * 1024 + s]);
    red[r * 8 + part] = mx; __syncthreads();
    if (tid < 16) { float v = red[tid * 8];
      for (int p = 1; p < 8; ++p) v = fmaxf(v, red[tid * 8 + p]); rmax[tid] = v; }
    __syncthreads();
    float m2 = rmax[r], s1 = 0.f, s2 = 0.f;
    for (int s = part * 128; s < part * 128 + 128; ++s) {
      float v = Sl[r * 1024 + s] - m2;
      s1 += expf(v); s2 += expf(v / SCALE);
    }
    red[r * 8 + part] = s1; __syncthreads();
    if (tid < 16) { float v = 0; for (int p = 0; p < 8; ++p) v += red[tid * 8 + p]; rsum1[tid] = v; }
    __syncthreads();
    red[r * 8 + part] = s2; __syncthreads();
    if (tid < 16) { float v = 0; for (int p = 0; p < 8; ++p) v += red[tid * 8 + p]; rsum2[tid] = v; }
    __syncthreads();
    if (tid < 16) {
      int t = t0 + tid;
      float pdiag = expf(Sl[tid * 1024 + t] - rmax[tid]) / rsum1[tid];
      att[b * T_ + t] = (1.f - pdiag) / SCALE;
    }
  }
  __syncthreads();

  // phase 3a: probabilities softmax(S/scale) as f16
  for (int i = tid; i < 16 * 1024; i += 128) {
    int r = i >> 10;
    P16[i] = (h16)(expf((Sl[i] - rmax[r]) / SCALE) / rsum2[r]);
  }
  __syncthreads();

  // phase 3b: out rows = P(16x1024) @ value(1024xDP), value stored [d][t]
  const h16* vTb = valueT + (size_t)b * DP * T_;
  for (int nt = 0; nt < 8; ++nt) {
    int n0 = wave * 128 + nt * 16;
    v8f acc = {};
    for (int k0 = 0; k0 < T_; k0 += 32) {
      v16h a = frag_a((const h16*)P16, 1024, k0, lane);
      v16h bb = frag_bt(vTb + (size_t)n0 * T_, T_, k0, lane);
      acc = WMMA(a, bb, acc);
    }
    int n = n0 + lo;
    if (n < D_) {
      float mn = mvec[b * D_ + n];
#pragma unroll
      for (int j = 0; j < 8; ++j) {
        int t = t0 + j + 8 * hi;
        out[((size_t)b * T_ + t) * D_ + n] = mn * acc[j];
      }
    }
  }
}

// ---------------- host launch ----------------------------------------------
extern "C" void kernel_launch(void* const* d_in, const int* in_sizes, int n_in,
                              void* d_out, int out_size, void* d_ws, size_t ws_size,
                              hipStream_t stream) {
  (void)in_sizes; (void)n_in; (void)out_size; (void)ws_size;
  const int*   words    = (const int*)d_in[0];
  const int*   pos      = (const int*)d_in[2];
  const int*   ner      = (const int*)d_in[3];
  const int*   subj_pos = (const int*)d_in[4];
  const int*   obj_pos  = (const int*)d_in[5];
  const int*   chunks   = (const int*)d_in[6];
  const int*   on_path  = (const int*)d_in[7];
  const float* dep_feat = (const float*)d_in[8];
  const float* emb_w    = (const float*)d_in[9];
  const float* pos_w    = (const float*)d_in[10];
  const float* ner_w    = (const float*)d_in[11];
  const float* chunk_w  = (const float*)d_in[12];
  const float* posi_w   = (const float*)d_in[13];
  const float* Wq_w = (const float*)d_in[14]; const float* Wq_b = (const float*)d_in[15];
  const float* Wc_w = (const float*)d_in[16]; const float* Wc_b = (const float*)d_in[17];
  const float* Wk_w = (const float*)d_in[18]; const float* Wk_b = (const float*)d_in[19];
  const float* Wm_w = (const float*)d_in[20]; const float* Wm_b = (const float*)d_in[21];
  const float* K_w  = (const float*)d_in[22]; const float* K_b  = (const float*)d_in[23];
  const float* Q_w  = (const float*)d_in[24]; const float* Q_b  = (const float*)d_in[25];
  const float* V_w  = (const float*)d_in[26]; const float* V_b  = (const float*)d_in[27];

  float* outp = (float*)d_out;                 // (B,T,500)
  float* attp = outp + (size_t)NROW * D_;      // (B,T)

  // workspace carve-out (~68 MB)
  char* ws = (char*)d_ws; size_t off = 0;
  auto carve = [&](size_t bytes) { void* p = ws + off; off += (bytes + 255) & ~(size_t)255; return p; };
  h16* g16   = (h16*)carve((size_t)NROW * DP * 2);
  h16* key16 = (h16*)carve((size_t)NROW * DP * 2);
  h16* qry16 = (h16*)carve((size_t)NROW * DP * 2);
  h16* valT  = (h16*)carve((size_t)NROW * DP * 2);    // [b][d][t]
  h16* KwT   = (h16*)carve((size_t)DP * DP * 2);
  h16* QwT   = (h16*)carve((size_t)DP * DP * 2);
  h16* VwT   = (h16*)carve((size_t)DP * DP * 2);
  float* subj = (float*)carve((size_t)B_ * D_ * 4);
  float* qv   = (float*)carve((size_t)B_ * D_ * 4);
  float* tv   = (float*)carve((size_t)B_ * D_ * 4);
  float* cv   = (float*)carve((size_t)B_ * D_ * 4);
  float* mv   = (float*)carve((size_t)B_ * D_ * 4);
  float* klog = (float*)carve((size_t)NROW * 4);
  float* ksm  = (float*)carve((size_t)NROW * 4);

  // 1. embeddings
  embed_kernel<<<(NROW * DP) / 256, 256, 0, stream>>>(words, pos, ner, subj_pos, obj_pos,
      chunks, on_path, dep_feat, emb_w, pos_w, ner_w, chunk_w, posi_w, g16);
  // 2. weight transposes
  wtrans_kernel<<<(DP * DP) / 256, 256, 0, stream>>>(K_w, KwT);
  wtrans_kernel<<<(DP * DP) / 256, 256, 0, stream>>>(Q_w, QwT);
  wtrans_kernel<<<(DP * DP) / 256, 256, 0, stream>>>(V_w, VwT);
  // 3. K/Q/V projections (WMMA)
  proj_kernel<<<dim3(8, 128), 256, 0, stream>>>(g16, KwT, K_b, key16, 0);
  proj_kernel<<<dim3(8, 128), 256, 0, stream>>>(g16, QwT, Q_b, qry16, 0);
  proj_kernel<<<dim3(8, 128), 256, 0, stream>>>(g16, VwT, V_b, valT, 1);
  // 4. pooling + attention-pooling MLP chain
  pool_kernel<<<(B_ * D_ + 255) / 256, 256, 0, stream>>>(subj_pos, g16, subj);
  q_kernel  <<<(B_ * D_ + 255) / 256, 256, 0, stream>>>(subj, Wq_w, Wq_b, qv);
  tv_kernel <<<(B_ * D_ + 255) / 256, 256, 0, stream>>>(qv, Wc_w, Wc_b, tv);
  klog_kernel<<<NROW / 256, 256, 0, stream>>>(tv, g16, Wk_w, Wk_b, klog);
  ksm_kernel<<<B_, 256, 0, stream>>>(klog, ksm);
  c_kernel  <<<(B_ * D_ + 255) / 256, 256, 0, stream>>>(ksm, g16, cv);
  m_kernel  <<<(B_ * D_ + 255) / 256, 256, 0, stream>>>(cv, subj, Wm_w, Wm_b, mv);
  // 5. fused attention (WMMA S-GEMM + softmax + att + WMMA P@V + m*out)
  attn_kernel<<<dim3(T_ / 16, B_), 128, 0, stream>>>(key16, qry16, valT, mv, outp, attp);
}
